// TimmPnPNystromAttention_20255065768199
// MI455X (gfx1250) — matrix operand order repaired and checked by reading, not compiled, not run on hardware
//
#include <hip/hip_runtime.h>
#include <hip/hip_bf16.h>
#include <math.h>

// ---------------------------------------------------------------------------
// CDNA5 WMMA types (wave32, v_wmma_f32_16x16x32_bf16)
// ---------------------------------------------------------------------------
typedef __bf16 bf16_t;
typedef __attribute__((ext_vector_type(16))) __bf16 v16bf;
typedef __attribute__((ext_vector_type(8)))  float  v8f;

union AFrag { v16bf v; bf16_t e[16]; uint4 q[2]; };

// A fragment 16x32 (MxK), row-major, lda multiple of 8, k0 multiple of 32,
// base 16B-aligned. No bounds checks: callers guarantee padded operands.
// Lane 0-15 : row = row0+lane,    K = k0 + {0..7, 16..23}
// Lane 16-31: row = row0+lane-16, K = k0 + {8..15, 24..31}
__device__ inline v16bf load_a_frag(const bf16_t* __restrict__ A, int lda,
                                    int row0, int k0) {
  int lane = threadIdx.x & 31;
  const bf16_t* p = A + (size_t)(row0 + (lane & 15)) * lda + k0 + ((lane >> 4) << 3);
  AFrag f;
  f.q[0] = *(const uint4*)p;         // K = kb + 0..7
  f.q[1] = *(const uint4*)(p + 16);  // K = kb + 16..23
  return f.v;
}

// B fragment 32x16 (KxN) from a TRANSPOSED store (N x K row-major, ldt mult 8).
// Lane 0-15 : col = n0+lane,    K = k0 + {0..7, 16..23}
// Lane 16-31: col = n0+lane-16, K = k0 + {8..15, 24..31}
__device__ inline v16bf load_bt_frag(const bf16_t* __restrict__ Bt, int ldt,
                                     int k0, int n0) {
  int lane = threadIdx.x & 31;
  const bf16_t* p = Bt + (size_t)(n0 + (lane & 15)) * ldt + k0 + ((lane >> 4) << 3);
  AFrag f;
  f.q[0] = *(const uint4*)p;         // K = kb + 0..7
  f.q[1] = *(const uint4*)(p + 16);  // K = kb + 16..23
  return f.v;
}

__device__ inline v8f wmma_bf16(v16bf a, v16bf b, v8f c) {
  return __builtin_amdgcn_wmma_f32_16x16x32_bf16(false, a, false, b,
                                                 (short)0, c, false, false);
}

// ---------------------------------------------------------------------------
// Problem constants + padded layouts
// ---------------------------------------------------------------------------
#define BB   32
#define NN   1025
#define CC   768
#define HH   12
#define DH   64
#define LL   65
#define BHZ  (BB*HH)     // 384
#define MROW (BB*NN)     // 32800 (= 1025*32, exact)
#define K3   (3*CC)      // 2304
#define NP   1040        // 1025 rows padded to 16
#define VP   1056        // 1025 padded to 32-multiple (K pad, zeroed)
#define LP   80          // 65 padded to 16 (M/N pad)
#define LK   96          // 65 padded to 32 (K pad, zeroed)
#define SCLD 1040        // S_C f32 leading dim

// ---------------------------------------------------------------------------
// Conversions / pad zeroing
// ---------------------------------------------------------------------------
__global__ void k_cvt_bf16(const float* __restrict__ src, bf16_t* __restrict__ dst,
                           long long n) {
  long long i = (long long)blockIdx.x * blockDim.x + threadIdx.x;
  long long stride = (long long)gridDim.x * blockDim.x;
  for (; i < n; i += stride) dst[i] = (bf16_t)src[i];
}

// f32 KxN -> bf16 NxK (transposed) for weight B-operands
__global__ void k_cvt_t_bf16(const float* __restrict__ src, bf16_t* __restrict__ dst,
                             int K, int N) {
  long long total = (long long)K * N;
  long long i = (long long)blockIdx.x * blockDim.x + threadIdx.x;
  long long stride = (long long)gridDim.x * blockDim.x;
  for (; i < total; i += stride) {
    int k = (int)(i / N), n = (int)(i - (long long)k * N);
    dst[(size_t)n * K + k] = (bf16_t)src[i];
  }
}

// zero rows 1025..1039 of (BHZ, NP, 64) q/k buffers
__global__ void k_pad_qk(bf16_t* __restrict__ p) {
  int z = blockIdx.x, t = threadIdx.x;  // 64 threads
  for (int r = NN; r < NP; ++r) p[((size_t)z * NP + r) * DH + t] = (bf16_t)0.0f;
}

// zero rows 65..79 of (BHZ, LP, 64) landmark buffers
__global__ void k_pad_m(bf16_t* __restrict__ p) {
  int z = blockIdx.x, t = threadIdx.x;  // 64 threads
  for (int r = LL; r < LP; ++r) p[((size_t)z * LP + r) * DH + t] = (bf16_t)0.0f;
}

// v_aug^T layout: (BHZ, LP rows = aug-dim, VP cols = tokens)
// row 64 = ones (tokens < 1025), rows 65..79 = 0, cols >= 1025 of rows 0..64 = 0
__global__ void k_vaugt_pad(bf16_t* __restrict__ vt) {
  int z = blockIdx.x;
  bf16_t* V = vt + (size_t)z * LP * VP;
  for (int i = threadIdx.x; i < LP * VP; i += blockDim.x) {
    int r = i / VP, c = i - r * VP;
    if (r >= LL)            V[i] = (bf16_t)0.0f;
    else if (r == DH)       V[i] = (c < NN) ? (bf16_t)1.0f : (bf16_t)0.0f;
    else if (c >= NN)       V[i] = (bf16_t)0.0f;
    // r<64, c<1025 written by QKV scatter
  }
}

// ---------------------------------------------------------------------------
// QKV GEMM: (32800x768)@(768x2304)+bias, 32x64 tile per wave (2x4 WMMA tiles).
// Fused scatter: q*scale -> qb, k -> kbf, v -> vaug^T. No predication.
// ---------------------------------------------------------------------------
__global__ void k_qkv_gemm(const bf16_t* __restrict__ xb,
                           const bf16_t* __restrict__ wt,   // 2304x768 (N x K)
                           const float* __restrict__ bias,
                           bf16_t* __restrict__ qb, bf16_t* __restrict__ kb,
                           bf16_t* __restrict__ vaugt) {
  int n0 = blockIdx.x * 64, m0 = blockIdx.y * 32;
  v8f acc[2][4] = {};
  for (int kk = 0; kk < CC; kk += 32) {
    if (kk + 32 < CC)
      __builtin_prefetch(&xb[(size_t)(m0 + (threadIdx.x & 31)) * CC + kk + 32], 0, 1);
    v16bf a0 = load_a_frag(xb, CC, m0, kk);
    v16bf a1 = load_a_frag(xb, CC, m0 + 16, kk);
#pragma unroll
    for (int nt = 0; nt < 4; ++nt) {
      v16bf b = load_bt_frag(wt, CC, kk, n0 + nt * 16);
      acc[0][nt] = wmma_bf16(a0, b, acc[0][nt]);
      acc[1][nt] = wmma_bf16(a1, b, acc[1][nt]);
    }
  }
  int lane = threadIdx.x & 31;
#pragma unroll
  for (int mt2 = 0; mt2 < 2; ++mt2) {
    int mbase = m0 + mt2 * 16 + ((lane >> 4) << 3);
#pragma unroll
    for (int nt = 0; nt < 4; ++nt) {
      int n = n0 + nt * 16 + (lane & 15);
      int which = n / CC, rem = n % CC, h = rem >> 6, d = rem & 63;
      float bsv = bias[n];
#pragma unroll
      for (int r = 0; r < 8; ++r) {
        int m = mbase + r;
        int bidx = m / NN, nn = m - bidx * NN;
        size_t z = (size_t)(bidx * HH + h);
        float val = acc[mt2][nt][r] + bsv;
        if (which == 0)      qb[(z * NP + nn) * DH + d] = (bf16_t)(val * 0.125f);
        else if (which == 1) kb[(z * NP + nn) * DH + d] = (bf16_t)val;
        else                 vaugt[((size_t)z * LP + d) * VP + nn] = (bf16_t)val;
      }
    }
  }
}

// ---------------------------------------------------------------------------
// Landmark pooling: (BHZ, NP, 64) -> (BHZ, LP, 64); cls row + 4x4 mean pool
// ---------------------------------------------------------------------------
__global__ void k_pool(const bf16_t* __restrict__ src, bf16_t* __restrict__ dst) {
  int z = blockIdx.y, l = blockIdx.x, d = threadIdx.x;  // d 0..63
  const bf16_t* S = src + (size_t)z * NP * DH;
  bf16_t out;
  if (l == 0) {
    out = S[d];
  } else {
    int i = (l - 1) >> 3, j = (l - 1) & 7;
    float s = 0.0f;
#pragma unroll
    for (int a = 0; a < 4; ++a)
#pragma unroll
      for (int b2 = 0; b2 < 4; ++b2) {
        int n = 1 + (i * 4 + a) * 32 + (j * 4 + b2);
        s += (float)S[(size_t)n * DH + d];
      }
    out = (bf16_t)(s * 0.0625f);
  }
  dst[((size_t)z * LP + l) * DH + d] = out;
}

// ---------------------------------------------------------------------------
// S_A = q @ k_m^T fused with row-max/exp -> expA (BHZ, NP, LK), K-pad zeroed
// ---------------------------------------------------------------------------
__global__ void k_sa_expA(const bf16_t* __restrict__ qb,
                          const bf16_t* __restrict__ km,
                          bf16_t* __restrict__ expA) {
  int mt = blockIdx.x, z = blockIdx.y;
  const bf16_t* Q  = qb + (size_t)z * NP * DH;
  const bf16_t* Km = km + (size_t)z * LP * DH;
  __shared__ float sc[16][80];
  v8f acc[5] = {};
  for (int kk = 0; kk < DH; kk += 32) {
    v16bf a = load_a_frag(Q, DH, mt * 16, kk);
#pragma unroll
    for (int nt = 0; nt < 5; ++nt) {
      v16bf b = load_bt_frag(Km, DH, kk, nt * 16);
      acc[nt] = wmma_bf16(a, b, acc[nt]);
    }
  }
  int lane = threadIdx.x & 31;
  int col = lane & 15, mb = (lane >> 4) << 3;
#pragma unroll
  for (int nt = 0; nt < 5; ++nt)
#pragma unroll
    for (int r = 0; r < 8; ++r) sc[mb + r][nt * 16 + col] = acc[nt][r];
  __syncthreads();
  if (lane < 16) {
    int gm = mt * 16 + lane;
    bf16_t* Orow = expA + ((size_t)z * NP + gm) * LK;
    if (gm < NN) {
      float mx = -1e30f;
      for (int c = 0; c < LL; ++c) mx = fmaxf(mx, sc[lane][c]);
      for (int c = 0; c < LL; ++c) Orow[c] = (bf16_t)__expf(sc[lane][c] - mx);
      for (int c = LL; c < LK; ++c) Orow[c] = (bf16_t)0.0f;   // K pad
    } else {
      for (int c = 0; c < LK; ++c) Orow[c] = (bf16_t)0.0f;    // M pad row
    }
  }
}

// ---------------------------------------------------------------------------
// S_C = q_m @ k^T -> f32 (BHZ, LP, SCLD); pads become zeros automatically
// ---------------------------------------------------------------------------
__global__ void k_sc_gemm(const bf16_t* __restrict__ qm,
                          const bf16_t* __restrict__ kb,
                          float* __restrict__ SC) {
  int nt = blockIdx.x, mt = blockIdx.y, z = blockIdx.z;
  const bf16_t* Qm = qm + (size_t)z * LP * DH;
  const bf16_t* Kk = kb + (size_t)z * NP * DH;
  v8f acc = {};
  for (int kk = 0; kk < DH; kk += 32) {
    v16bf a = load_a_frag(Qm, DH, mt * 16, kk);
    v16bf b = load_bt_frag(Kk, DH, kk, nt * 16);
    acc = wmma_bf16(a, b, acc);
  }
  int lane = threadIdx.x & 31;
  int n = nt * 16 + (lane & 15);
  int mb = mt * 16 + ((lane >> 4) << 3);
  float* O = SC + (size_t)z * LP * SCLD;
#pragma unroll
  for (int r = 0; r < 8; ++r) O[(size_t)(mb + r) * SCLD + n] = acc[r];
}

__global__ void k_rowmax(const float* __restrict__ SC, float* __restrict__ maxC) {
  int row = blockIdx.x * blockDim.x + threadIdx.x;
  if (row < BHZ * LL) {
    int z = row / LL, r = row - z * LL;
    const float* R = SC + ((size_t)z * LP + r) * SCLD;
    float mx = -1e30f;
    for (int c = 0; c < NN; ++c) mx = fmaxf(mx, R[c]);
    maxC[row] = mx;
  }
}

// expC (BHZ, LK rows, VP cols): valid 65x1025, all pads zeroed here
__global__ void k_expC(const float* __restrict__ SC, const float* __restrict__ maxC,
                       bf16_t* __restrict__ expC) {
  int zr = blockIdx.y;
  int z = zr / LK, r = zr - z * LK;
  int c = blockIdx.x * blockDim.x + threadIdx.x;
  if (c < VP) {
    float v = 0.0f;
    if (r < LL && c < NN)
      v = __expf(SC[((size_t)z * LP + r) * SCLD + c] - maxC[z * LL + r]);
    expC[((size_t)z * LK + r) * VP + c] = (bf16_t)v;
  }
}

// ---------------------------------------------------------------------------
// S_B/expB (f32, tiny) then Moore-Penrose pinv in LDS -> pinvb (LK x LK, padded)
// ---------------------------------------------------------------------------
__global__ void k_sb_expB(const bf16_t* __restrict__ qm,
                          const bf16_t* __restrict__ km,
                          const float* __restrict__ maxC,
                          float* __restrict__ expB) {
  int z = blockIdx.x;
  const bf16_t* Q = qm + (size_t)z * LP * DH;
  const bf16_t* K = km + (size_t)z * LP * DH;
  const float* mc = maxC + (size_t)z * LL;
  float* O = expB + (size_t)z * LL * LL;
  for (int i = threadIdx.x; i < LL * LL; i += blockDim.x) {
    int r = i / LL, c = i - r * LL;
    float s = 0.0f;
    for (int d = 0; d < DH; ++d) s += (float)Q[r * DH + d] * (float)K[c * DH + d];
    O[i] = __expf(fmaxf(s - mc[r], -88.0f));
  }
}

__global__ void k_pinv(const float* __restrict__ expB, bf16_t* __restrict__ pinvb) {
  int z = blockIdx.x;
  const float* X = expB + (size_t)z * LL * LL;
  __shared__ float  Zs[LL * LL], Ts[LL * LL], Qs[LL * LL];
  __shared__ bf16_t Ps[LL * LL];
  __shared__ float  rs[LL], cs[LL], scl;
  int t = threadIdx.x;
  if (t < LL) {
    float s = 0.0f, s2 = 0.0f;
    for (int c = 0; c < LL; ++c) { s += fabsf(X[t * LL + c]); s2 += fabsf(X[c * LL + t]); }
    rs[t] = s; cs[t] = s2;
  }
  __syncthreads();
  if (t == 0) {
    float mr = 0.0f, mc2 = 0.0f;
    for (int i = 0; i < LL; ++i) { mr = fmaxf(mr, rs[i]); mc2 = fmaxf(mc2, cs[i]); }
    scl = 1.0f / (mr * mc2 + 1e-15f);
  }
  __syncthreads();
  for (int i = t; i < LL * LL; i += blockDim.x) {
    int r = i / LL, c = i - r * LL;
    Zs[i] = X[c * LL + r] * scl;
  }
  __syncthreads();
  for (int it = 0; it < 6; ++it) {
    for (int i = t; i < LL * LL; i += blockDim.x) {    // P = X @ Z
      int r = i / LL, c = i - r * LL;
      float s = 0.0f;
      for (int k = 0; k < LL; ++k) s += X[r * LL + k] * Zs[k * LL + c];
      Ps[i] = (bf16_t)s;
    }
    __syncthreads();
    for (int i = t; i < LL * LL; i += blockDim.x) {    // T = 7I - P
      int r = i / LL, c = i - r * LL;
      Ts[i] = (r == c ? 7.0f : 0.0f) - (float)Ps[i];
    }
    __syncthreads();
    for (int i = t; i < LL * LL; i += blockDim.x) {    // Q = 15I - P@T
      int r = i / LL, c = i - r * LL;
      float s = 0.0f;
      for (int k = 0; k < LL; ++k) s += (float)Ps[r * LL + k] * Ts[k * LL + c];
      Qs[i] = (r == c ? 15.0f : 0.0f) - s;
    }
    __syncthreads();
    for (int i = t; i < LL * LL; i += blockDim.x) {    // T = 13I - P@Q
      int r = i / LL, c = i - r * LL;
      float s = 0.0f;
      for (int k = 0; k < LL; ++k) s += (float)Ps[r * LL + k] * Qs[k * LL + c];
      Ts[i] = (r == c ? 13.0f : 0.0f) - s;
    }
    __syncthreads();
    for (int i = t; i < LL * LL; i += blockDim.x) {    // Q = 0.25 * Z@T
      int r = i / LL, c = i - r * LL;
      float s = 0.0f;
      for (int k = 0; k < LL; ++k) s += Zs[r * LL + k] * Ts[k * LL + c];
      Qs[i] = 0.25f * s;
    }
    __syncthreads();
    for (int i = t; i < LL * LL; i += blockDim.x) Zs[i] = Qs[i];
    __syncthreads();
  }
  bf16_t* O = pinvb + (size_t)z * LK * LK;            // LK x LK, pads zeroed
  for (int i = t; i < LK * LK; i += blockDim.x) {
    int r = i / LK, c = i - r * LK;
    O[i] = (r < LL && c < LL) ? (bf16_t)Zs[r * LL + c] : (bf16_t)0.0f;
  }
}

// ---------------------------------------------------------------------------
// T1 = expC @ v_aug, stored TRANSPOSED (LP x LK): rows n, cols m (K of next op)
// ---------------------------------------------------------------------------
__global__ void k_t1(const bf16_t* __restrict__ expC, const bf16_t* __restrict__ vaugt,
                     bf16_t* __restrict__ T1t) {
  int nt = blockIdx.x, mt = blockIdx.y, z = blockIdx.z;
  const bf16_t* A  = expC + (size_t)z * LK * VP;
  const bf16_t* Bv = vaugt + (size_t)z * LP * VP;
  v8f acc = {};
  for (int kk = 0; kk < NN; kk += 32) {
    v16bf a = load_a_frag(A, VP, mt * 16, kk);
    v16bf b = load_bt_frag(Bv, VP, kk, nt * 16);
    acc = wmma_bf16(a, b, acc);
  }
  int lane = threadIdx.x & 31;
  int n = nt * 16 + (lane & 15);
  int mb = mt * 16 + ((lane >> 4) << 3);
  bf16_t* O = T1t + (size_t)z * LP * LK;
#pragma unroll
  for (int r = 0; r < 8; ++r) O[(size_t)n * LK + mb + r] = (bf16_t)acc[r];
}

// T2 = pinv @ T1, stored TRANSPOSED (LP x LK)
__global__ void k_t2(const bf16_t* __restrict__ pinvb, const bf16_t* __restrict__ T1t,
                     bf16_t* __restrict__ T2t) {
  int nt = blockIdx.x, mt = blockIdx.y, z = blockIdx.z;
  const bf16_t* A  = pinvb + (size_t)z * LK * LK;
  const bf16_t* Bm = T1t + (size_t)z * LP * LK;
  v8f acc = {};
  for (int kk = 0; kk < LL; kk += 32) {
    v16bf a = load_a_frag(A, LK, mt * 16, kk);
    v16bf b = load_bt_frag(Bm, LK, kk, nt * 16);
    acc = wmma_bf16(a, b, acc);
  }
  int lane = threadIdx.x & 31;
  int n = nt * 16 + (lane & 15);
  int mb = mt * 16 + ((lane >> 4) << 3);
  bf16_t* O = T2t + (size_t)z * LP * LK;
#pragma unroll
  for (int r = 0; r < 8; ++r) O[(size_t)n * LK + mb + r] = (bf16_t)acc[r];
}

// ---------------------------------------------------------------------------
// prod = expA @ T2; ctx = prod[:, :64] / max(prod[:,64], 1e-8), fused with
// (B,H,N,Dh) -> (B,N,C) transpose. bf16 output for the final GEMM.
// ---------------------------------------------------------------------------
__global__ void k_prod_ctx(const bf16_t* __restrict__ expA, const bf16_t* __restrict__ T2t,
                           bf16_t* __restrict__ ctx) {
  int mt = blockIdx.x, z = blockIdx.y;
  int b = z / HH, h = z - b * HH;
  const bf16_t* A  = expA + (size_t)z * NP * LK;
  const bf16_t* Bm = T2t + (size_t)z * LP * LK;
  __shared__ float sc[16][80];
  v8f acc[5] = {};
  for (int kk = 0; kk < LL; kk += 32) {
    v16bf a = load_a_frag(A, LK, mt * 16, kk);
#pragma unroll
    for (int nt = 0; nt < 5; ++nt) {
      v16bf bb = load_bt_frag(Bm, LK, kk, nt * 16);
      acc[nt] = wmma_bf16(a, bb, acc[nt]);
    }
  }
  int lane = threadIdx.x & 31;
  int col = lane & 15, mb = (lane >> 4) << 3;
#pragma unroll
  for (int nt = 0; nt < 5; ++nt)
#pragma unroll
    for (int r = 0; r < 8; ++r) sc[mb + r][nt * 16 + col] = acc[nt][r];
  __syncthreads();
  if (lane < 16) {
    int gm = mt * 16 + lane;
    if (gm < NN) {
      float inv = 1.0f / fmaxf(sc[lane][DH], 1e-8f);
      bf16_t* O = ctx + ((size_t)b * NN + gm) * CC + h * DH;
      for (int d = 0; d < DH; ++d) O[d] = (bf16_t)(sc[lane][d] * inv);
    }
  }
}

// ---------------------------------------------------------------------------
// Output projection: ctx (32800x768) @ proj_w + bias -> f32, 32x64 tile/wave
// ---------------------------------------------------------------------------
__global__ void k_proj_gemm(const bf16_t* __restrict__ ctx, const bf16_t* __restrict__ wt,
                            const float* __restrict__ bias, float* __restrict__ out) {
  int n0 = blockIdx.x * 64, m0 = blockIdx.y * 32;
  v8f acc[2][4] = {};
  for (int kk = 0; kk < CC; kk += 32) {
    if (kk + 32 < CC)
      __builtin_prefetch(&ctx[(size_t)(m0 + (threadIdx.x & 31)) * CC + kk + 32], 0, 1);
    v16bf a0 = load_a_frag(ctx, CC, m0, kk);
    v16bf a1 = load_a_frag(ctx, CC, m0 + 16, kk);
#pragma unroll
    for (int nt = 0; nt < 4; ++nt) {
      v16bf b = load_bt_frag(wt, CC, kk, n0 + nt * 16);
      acc[0][nt] = wmma_bf16(a0, b, acc[0][nt]);
      acc[1][nt] = wmma_bf16(a1, b, acc[1][nt]);
    }
  }
  int lane = threadIdx.x & 31;
#pragma unroll
  for (int mt2 = 0; mt2 < 2; ++mt2) {
    int mb = m0 + mt2 * 16 + ((lane >> 4) << 3);
#pragma unroll
    for (int nt = 0; nt < 4; ++nt) {
      int n = n0 + nt * 16 + (lane & 15);
      float bs = bias[n];
#pragma unroll
      for (int r = 0; r < 8; ++r)
        out[(size_t)(mb + r) * CC + n] = acc[mt2][nt][r] + bs;
    }
  }
}

// ---------------------------------------------------------------------------
// Host launcher
// ---------------------------------------------------------------------------
extern "C" void kernel_launch(void* const* d_in, const int* in_sizes, int n_in,
                              void* d_out, int out_size, void* d_ws, size_t ws_size,
                              hipStream_t stream) {
  (void)in_sizes; (void)n_in; (void)out_size; (void)ws_size;
  const float* x      = (const float*)d_in[0];
  const float* qkv_w  = (const float*)d_in[1];
  const float* qkv_b  = (const float*)d_in[2];
  const float* proj_w = (const float*)d_in[3];
  const float* proj_b = (const float*)d_in[4];
  float* out = (float*)d_out;

  char* w = (char*)d_ws;
  size_t off = 0;
  auto carve = [&](size_t bytes) -> char* {
    char* p = w + off;
    off = (off + bytes + 255) & ~(size_t)255;
    return p;
  };
  bf16_t* xb     = (bf16_t*)carve((size_t)MROW * CC * 2);
  bf16_t* wqkvt  = (bf16_t*)carve((size_t)K3 * CC * 2);   // 2304 x 768 (NxK)
  bf16_t* wprojt = (bf16_t*)carve((size_t)CC * CC * 2);   // 768 x 768  (NxK)
  bf16_t* qb     = (bf16_t*)carve((size_t)BHZ * NP * DH * 2);
  bf16_t* kbf    = (bf16_t*)carve((size_t)BHZ * NP * DH * 2);
  bf16_t* vaugt  = (bf16_t*)carve((size_t)BHZ * LP * VP * 2);
  bf16_t* qm     = (bf16_t*)carve((size_t)BHZ * LP * DH * 2);
  bf16_t* km     = (bf16_t*)carve((size_t)BHZ * LP * DH * 2);
  bf16_t* expA   = (bf16_t*)carve((size_t)BHZ * NP * LK * 2);
  float*  SC     = (float*) carve((size_t)BHZ * LP * SCLD * 4);
  float*  maxC   = (float*) carve((size_t)BHZ * LL * 4);
  bf16_t* expC   = (bf16_t*)carve((size_t)BHZ * LK * VP * 2);
  float*  expB   = (float*) carve((size_t)BHZ * LL * LL * 4);
  bf16_t* pinvb  = (bf16_t*)carve((size_t)BHZ * LK * LK * 2);
  bf16_t* T1t    = (bf16_t*)carve((size_t)BHZ * LP * LK * 2);
  bf16_t* T2t    = (bf16_t*)carve((size_t)BHZ * LP * LK * 2);
  bf16_t* ctx    = (bf16_t*)carve((size_t)MROW * CC * 2);

  // conversions (weights transposed for b128 B-fragment loads)
  k_cvt_bf16<<<2048, 256, 0, stream>>>(x, xb, (long long)MROW * CC);
  k_cvt_t_bf16<<<512, 256, 0, stream>>>(qkv_w, wqkvt, CC, K3);
  k_cvt_t_bf16<<<256, 256, 0, stream>>>(proj_w, wprojt, CC, CC);

  // zero K/M pads (ws is poisoned, zeros required every call)
  k_pad_qk<<<BHZ, 64, 0, stream>>>(qb);
  k_pad_qk<<<BHZ, 64, 0, stream>>>(kbf);
  k_vaugt_pad<<<BHZ, 256, 0, stream>>>(vaugt);

  k_qkv_gemm<<<dim3(K3 / 64, MROW / 32), 32, 0, stream>>>(xb, wqkvt, qkv_b,
                                                          qb, kbf, vaugt);
  k_pool<<<dim3(LL, BHZ), 64, 0, stream>>>(qb, qm);
  k_pool<<<dim3(LL, BHZ), 64, 0, stream>>>(kbf, km);
  k_pad_m<<<BHZ, 64, 0, stream>>>(qm);
  k_pad_m<<<BHZ, 64, 0, stream>>>(km);

  k_sa_expA<<<dim3(NP / 16, BHZ), 32, 0, stream>>>(qb, km, expA);
  k_sc_gemm<<<dim3(SCLD / 16, LP / 16, BHZ), 32, 0, stream>>>(qm, kbf, SC);
  k_rowmax<<<(BHZ * LL + 255) / 256, 256, 0, stream>>>(SC, maxC);
  k_expC<<<dim3((VP + 255) / 256, BHZ * LK), 256, 0, stream>>>(SC, maxC, expC);
  k_sb_expB<<<BHZ, 256, 0, stream>>>(qm, km, maxC, expB);
  k_pinv<<<BHZ, 256, 0, stream>>>(expB, pinvb);

  k_t1<<<dim3(LP / 16, LK / 16, BHZ), 32, 0, stream>>>(expC, vaugt, T1t);
  k_t2<<<dim3(LP / 16, LK / 16, BHZ), 32, 0, stream>>>(pinvb, T1t, T2t);
  k_prod_ctx<<<dim3(NP / 16, BHZ), 32, 0, stream>>>(expA, T2t, ctx);
  k_proj_gemm<<<dim3(CC / 64, MROW / 32), 32, 0, stream>>>(ctx, wprojt, proj_b, out);
}
